// Net_74388833566923
// MI455X (gfx1250) — compile-verified, matrix operand order checked by reading, not run on hardware
//
#include <hip/hip_runtime.h>

typedef _Float16 v8h  __attribute__((ext_vector_type(8)));
typedef _Float16 v16h __attribute__((ext_vector_type(16)));
typedef float    v8f  __attribute__((ext_vector_type(8)));

#define INF_    64
#define ECF     16
#define TSTEPS  4
#define NDEG    10
#define KPAD    96
#define NGRAPH  64
#define MTILE   64   // edges per block (4 sub-tiles of 16)

// ---------------- prep kernels ----------------

__global__ void k_init_deg(int* deg, int N) {
    int i = blockIdx.x * blockDim.x + threadIdx.x;
    if (i < N) deg[i] = 1;              // self loop
}

__global__ void k_init_small(int* counts, float* pooled) {
    int i = threadIdx.x;
    if (i < 16) counts[i] = 0;
    if (i < NGRAPH * 10) pooled[i] = 0.f;
}

__global__ void k_degree(const int* __restrict__ eidx, int* deg, int E) {
    int e = blockIdx.x * blockDim.x + threadIdx.x;
    if (e < E) atomicAdd(&deg[eidx[e]], 1);
}

__global__ void k_count(const int* __restrict__ eidx, const int* __restrict__ deg,
                        int* counts, int E, int Etot) {
    int e = blockIdx.x * blockDim.x + threadIdx.x;
    if (e >= Etot) return;
    int s = (e < E) ? eidx[e] : (e - E);
    int d = deg[s]; if (d > NDEG) d = NDEG;
    atomicAdd(&counts[d - 1], 1);
}

__global__ void k_scan(const int* counts, int* offsets, int* cursors) {
    if (threadIdx.x == 0) {
        int acc = 0;
        for (int i = 0; i < NDEG; ++i) { offsets[i] = acc; cursors[i] = acc; acc += counts[i]; }
        offsets[NDEG] = acc;
    }
}

__global__ void k_scatter(const int* __restrict__ eidx, const int* __restrict__ deg,
                          int* cursors, int* elist, int E, int Etot) {
    int e = blockIdx.x * blockDim.x + threadIdx.x;
    if (e >= Etot) return;
    int s = (e < E) ? eidx[e] : (e - E);
    int d = deg[s]; if (d > NDEG) d = NDEG;
    int pos = atomicAdd(&cursors[d - 1], 1);
    elist[pos] = e;
}

// W_msg [T,10,80,64] (row-major) -> Wt [T,10,64,KPAD] f16, transposed + zero-padded K
__global__ void k_convW(const float* __restrict__ W, _Float16* __restrict__ Wt, int total) {
    int i = blockIdx.x * blockDim.x + threadIdx.x;
    if (i >= total) return;
    int k   = i % KPAD;
    int rem = i / KPAD;
    int n   = rem % INF_;
    int td  = rem / INF_;
    Wt[i] = (k < (INF_ + ECF)) ? (_Float16)W[(td * (INF_ + ECF) + k) * INF_ + n] : (_Float16)0.f;
}

// edge_attr -> f16, self-loop rows zero
__global__ void k_convE(const float* __restrict__ ea, _Float16* __restrict__ eah, int E, int Etot) {
    int i = blockIdx.x * blockDim.x + threadIdx.x;
    if (i >= Etot * ECF) return;
    eah[i] = (i < E * ECF) ? (_Float16)ea[i] : (_Float16)0.f;
}

// x fp32 -> f16, and zero the next accumulator buffer
__global__ void k_convX(const float* __restrict__ x, _Float16* __restrict__ xh,
                        float* __restrict__ xnext, int total) {
    int i = blockIdx.x * blockDim.x + threadIdx.x;
    if (i >= total) return;
    xh[i] = (_Float16)x[i];
    xnext[i] = 0.f;
}

__device__ __forceinline__ float fast_sigmoid(float x) {
    // 1 / (1 + exp(-x)) with hardware v_rcp_f32 (no IEEE div sequence)
    return __builtin_amdgcn_rcpf(1.f + __expf(-x));
}

// ---------------- WMMA message kernel ----------------
// block = 128 threads = 4 waves. Block tile: 64 edges x 64 outputs.
// wave w owns output columns [w*16, w*16+16); its B fragments are loaded ONCE
// and reused across 4 M-subtiles (12 WMMAs per wave per block).
// Invalid tile rows are clamped to a dump row (row index Ndump) so the
// scatter epilogue is fully branch-free (EXEC never changes).
__global__ void __launch_bounds__(128) k_msg(
    const _Float16* __restrict__ xh, const _Float16* __restrict__ eah,
    const _Float16* __restrict__ Wt, const int* __restrict__ elist,
    const int* __restrict__ offsets, const int* __restrict__ counts,
    const int* __restrict__ eidx, float* __restrict__ xout,
    int E, int t, int Ndump)
{
    int d    = blockIdx.y;
    int cnt  = counts[d];
    int tile = blockIdx.x;
    if (tile * MTILE >= cnt) return;         // uniform exit: EXEC stays full below

    __shared__ int s_src[MTILE];
    __shared__ int s_edge[MTILE];
    __shared__ int s_dst[MTILE];

    int tid = threadIdx.x;
    if (tid < MTILE) {
        int p = tile * MTILE + tid;
        if (p < cnt) {
            int e = elist[offsets[d] + p];
            s_edge[tid] = e;
            s_src[tid]  = (e < E) ? eidx[e]     : (e - E);
            s_dst[tid]  = (e < E) ? eidx[E + e] : (e - E);
        } else {
            s_edge[tid] = 0;      // clamp to valid memory, branch-free loads
            s_src[tid]  = 0;
            s_dst[tid]  = Ndump;  // dump row: written, never read
        }
    }
    __syncthreads();

    int lane  = tid & 31;
    int wave  = tid >> 5;                    // N-tile index (output cols wave*16..+15)
    int l15   = lane & 15;
    int koff  = (lane >> 4) ? 8 : 0;         // per-lane K sub-chunk
    int mbase = (lane >> 4) ? 8 : 0;         // C/D row base for this lane half

    // ---- B fragments: column n = wave*16 + l15 of W[t,d], K pattern {koff..+7, koff+16..+23}
    const _Float16* wbase = Wt + (((size_t)(t * NDEG + d) * INF_) + (wave * 16 + l15)) * KPAD + koff;
    v16h B0, B1, B2;
    {
        v8h b0 = *(const v8h*)(wbase);
        v8h b1 = *(const v8h*)(wbase + 16);
        B0 = __builtin_shufflevector(b0, b1, 0,1,2,3,4,5,6,7,8,9,10,11,12,13,14,15);
        v8h b2 = *(const v8h*)(wbase + 32);
        v8h b3 = *(const v8h*)(wbase + 48);
        B1 = __builtin_shufflevector(b2, b3, 0,1,2,3,4,5,6,7,8,9,10,11,12,13,14,15);
        v8h b4 = *(const v8h*)(wbase + 64);
        v8h b5 = *(const v8h*)(wbase + 80);  // zero-padded rows of Wt
        B2 = __builtin_shufflevector(b4, b5, 0,1,2,3,4,5,6,7,8,9,10,11,12,13,14,15);
    }

    int n = wave * 16 + l15;                 // output column this lane writes
    v8h z = {};

#pragma unroll
    for (int mt = 0; mt < MTILE / 16; ++mt) {
        int m = mt * 16 + l15;               // A row within block tile
        const _Float16* xrow = xh  + (size_t)s_src[m] * INF_;
        const _Float16* erow = eah + (size_t)s_edge[m] * ECF;

        v8f c = {};
        // K = 0..31 (x features)
        {
            v8h a0 = *(const v8h*)(xrow + koff);
            v8h a1 = *(const v8h*)(xrow + koff + 16);
            v16h A = __builtin_shufflevector(a0, a1, 0,1,2,3,4,5,6,7,8,9,10,11,12,13,14,15);
            c = __builtin_amdgcn_wmma_f32_16x16x32_f16(false, A, false, B0, (short)0, c, false, false);
        }
        // K = 32..63 (x features)
        {
            v8h a0 = *(const v8h*)(xrow + 32 + koff);
            v8h a1 = *(const v8h*)(xrow + 48 + koff);
            v16h A = __builtin_shufflevector(a0, a1, 0,1,2,3,4,5,6,7,8,9,10,11,12,13,14,15);
            c = __builtin_amdgcn_wmma_f32_16x16x32_f16(false, A, false, B1, (short)0, c, false, false);
        }
        // K = 64..79 edge_attr, K = 80..95 zero pad
        {
            v8h a0 = *(const v8h*)(erow + koff);
            v16h A = __builtin_shufflevector(a0, z, 0,1,2,3,4,5,6,7,8,9,10,11,12,13,14,15);
            c = __builtin_amdgcn_wmma_f32_16x16x32_f16(false, A, false, B2, (short)0, c, false, false);
        }

        // Fetch the 8 consecutive dst rows for this lane half in one go
        // (32B-aligned -> compiler can use ds_load_b128 pairs).
        int dsts[8];
#pragma unroll
        for (int r = 0; r < 8; ++r) dsts[r] = s_dst[mt * 16 + mbase + r];

        // sigmoid + unconditional scatter-add. C/D layout: vgpr r -> (M = r + mbase, N = l15)
#pragma unroll
        for (int r = 0; r < 8; ++r) {
            float v = fast_sigmoid(c[r]);
            atomicAdd(&xout[(size_t)dsts[r] * INF_ + n], v);
        }
    }
}

// ---------------- readout + pooling ----------------
__global__ void k_readout(const float* __restrict__ x, const float* __restrict__ Wr,
                          const int* __restrict__ batch, float* __restrict__ pooled, int N)
{
    int nidx = blockIdx.x * blockDim.x + threadIdx.x;
    if (nidx >= N) return;
    float xr[INF_];
#pragma unroll
    for (int i = 0; i < INF_; ++i) xr[i] = x[(size_t)nidx * INF_ + i];
    float acc[10];
#pragma unroll
    for (int j = 0; j < 10; ++j) acc[j] = 0.f;
    for (int t = 0; t < TSTEPS; ++t) {
        float o[10];
        for (int j = 0; j < 10; ++j) {
            float s = 0.f;
#pragma unroll
            for (int i = 0; i < INF_; ++i) s += xr[i] * Wr[(t * INF_ + i) * 10 + j];
            o[j] = s;
        }
        float mx = o[0];
        for (int j = 1; j < 10; ++j) mx = fmaxf(mx, o[j]);
        float se = 0.f;
        for (int j = 0; j < 10; ++j) { o[j] = __expf(o[j] - mx); se += o[j]; }
        float inv = __builtin_amdgcn_rcpf(se);
        for (int j = 0; j < 10; ++j) acc[j] += o[j] * inv;
    }
    int g = batch[nidx];
    for (int j = 0; j < 10; ++j) atomicAdd(&pooled[g * 10 + j], acc[j]);
}

// ---------------- tiny MLP head ----------------
__device__ __forceinline__ float lrelu(float v) { return v > 0.f ? v : 0.01f * v; }

__global__ void k_mlp(const float* __restrict__ pooled,
                      const float* __restrict__ w1, const float* __restrict__ b1,
                      const float* __restrict__ w2, const float* __restrict__ b2,
                      const float* __restrict__ w3, const float* __restrict__ b3,
                      float* __restrict__ out)
{
    int g = threadIdx.x;
    if (g >= NGRAPH) return;
    float p[10];
#pragma unroll
    for (int i = 0; i < 10; ++i) p[i] = pooled[g * 10 + i];
    float h2[64];
#pragma unroll
    for (int j = 0; j < 64; ++j) h2[j] = b2[j];
    for (int i = 0; i < 128; ++i) {
        float s = b1[i];
#pragma unroll
        for (int k = 0; k < 10; ++k) s += p[k] * w1[k * 128 + i];
        s = lrelu(s);
        for (int j = 0; j < 64; ++j) h2[j] += s * w2[i * 64 + j];
    }
    float s = b3[0];
    for (int j = 0; j < 64; ++j) s += lrelu(h2[j]) * w3[j];
    out[g] = lrelu(s);
}

// ---------------- host launcher ----------------
extern "C" void kernel_launch(void* const* d_in, const int* in_sizes, int n_in,
                              void* d_out, int out_size, void* d_ws, size_t ws_size,
                              hipStream_t stream) {
    const float* x     = (const float*)d_in[0];
    const float* ea    = (const float*)d_in[1];
    const float* Wmsg  = (const float*)d_in[2];
    const float* Wread = (const float*)d_in[3];
    const float* fc1w  = (const float*)d_in[4];
    const float* fc1b  = (const float*)d_in[5];
    const float* fc2w  = (const float*)d_in[6];
    const float* fc2b  = (const float*)d_in[7];
    const float* fc3w  = (const float*)d_in[8];
    const float* fc3b  = (const float*)d_in[9];
    const int*   eidx  = (const int*)d_in[10];
    const int*   batch = (const int*)d_in[11];

    int N    = in_sizes[0] / INF_;
    int E    = in_sizes[1] / ECF;
    int Etot = N + E;

    char* w = (char*)d_ws;
    auto carve = [&](size_t bytes) { char* p = w; w += (bytes + 255) & ~(size_t)255; return p; };
    int*       deg     = (int*)carve((size_t)N * 4);
    int*       counts  = (int*)carve(64);
    int*       offsets = (int*)carve(64);
    int*       cursors = (int*)carve(64);
    int*       elist   = (int*)carve((size_t)Etot * 4);
    float*     xbufA   = (float*)carve((size_t)(N + 1) * INF_ * 4);  // +1 dump row
    float*     xbufB   = (float*)carve((size_t)(N + 1) * INF_ * 4);  // +1 dump row
    _Float16*  xh      = (_Float16*)carve((size_t)N * INF_ * 2);
    _Float16*  eah     = (_Float16*)carve((size_t)Etot * ECF * 2);
    _Float16*  Wt      = (_Float16*)carve((size_t)TSTEPS * NDEG * INF_ * KPAD * 2);
    float*     pooled  = (float*)carve(NGRAPH * 10 * 4);

    k_init_deg  <<<(N + 255) / 256, 256, 0, stream>>>(deg, N);
    k_init_small<<<1, 640, 0, stream>>>(counts, pooled);
    k_degree    <<<(E + 255) / 256, 256, 0, stream>>>(eidx, deg, E);
    k_count     <<<(Etot + 255) / 256, 256, 0, stream>>>(eidx, deg, counts, E, Etot);
    k_scan      <<<1, 32, 0, stream>>>(counts, offsets, cursors);
    k_scatter   <<<(Etot + 255) / 256, 256, 0, stream>>>(eidx, deg, cursors, elist, E, Etot);

    int wtot = TSTEPS * NDEG * INF_ * KPAD;
    k_convW<<<(wtot + 255) / 256, 256, 0, stream>>>(Wmsg, Wt, wtot);
    k_convE<<<((Etot * ECF) + 255) / 256, 256, 0, stream>>>(ea, eah, E, Etot);

    const float* xcur = x;
    float* bufs[2] = {xbufA, xbufB};
    int tilesx = (Etot + MTILE - 1) / MTILE;
    for (int t = 0; t < TSTEPS; ++t) {
        float* xnext = bufs[t & 1];
        k_convX<<<((N * INF_) + 255) / 256, 256, 0, stream>>>(xcur, xh, xnext, N * INF_);
        dim3 grid(tilesx, NDEG);
        k_msg<<<grid, 128, 0, stream>>>(xh, eah, Wt, elist, offsets, counts, eidx, xnext, E, t, N);
        xcur = xnext;
    }

    k_readout<<<(N + 127) / 128, 128, 0, stream>>>(xcur, Wread, batch, pooled, N);
    k_mlp<<<1, 64, 0, stream>>>(pooled, fc1w, fc1b, fc2w, fc2b, fc3w, fc3b, (float*)d_out);
}
